// CausalSelfAttention_17626545782956
// MI455X (gfx1250) — matrix operand order repaired and checked by reading, not compiled, not run on hardware
//
#include <hip/hip_runtime.h>

typedef unsigned short u16;
typedef __attribute__((ext_vector_type(16))) __bf16 bf16x16;
typedef __attribute__((ext_vector_type(8)))  float  floatx8;
typedef __attribute__((ext_vector_type(4)))  unsigned int uintx4;
typedef __attribute__((ext_vector_type(4)))  float  floatx4;
typedef __attribute__((ext_vector_type(2)))  unsigned int uintx2;

#define D_MODEL 1024
#define SEQ     2048
#define BATCH   2
#define HEADS   16
#define DHEAD   64
#define M_TOT   (BATCH * SEQ)   // 4096

union Frag32B { bf16x16 v; uintx4 q[2]; u16 u[16]; };

__device__ __forceinline__ u16 f32_to_bf16(float f) {
  unsigned u = __float_as_uint(f);
  unsigned r = u + 0x7FFFu + ((u >> 16) & 1u);   // round-to-nearest-even
  return (u16)(r >> 16);
}

// CDNA5 async copy: global memory -> LDS, 16B per lane, tracked by ASYNCcnt.
__device__ __forceinline__ void async_copy_b128(u16* lds, const u16* gaddr) {
  asm volatile("global_load_async_to_lds_b128 %0, %1, off"
               :: "v"((unsigned)(unsigned long long)(size_t)lds),
                  "v"((unsigned long long)(size_t)gaddr)
               : "memory");
}
__device__ __forceinline__ void wait_async0() {
  asm volatile("s_wait_asynccnt 0" ::: "memory");
}

// ---------------------------------------------------------------- fp32 -> bf16
__global__ __launch_bounds__(256)
void cvt_f32_bf16_kernel(const float* __restrict__ in, u16* __restrict__ out, int n4) {
  int i = blockIdx.x * blockDim.x + threadIdx.x;
  if (i >= n4) return;
  floatx4 f = *(const floatx4*)(in + 4 * (size_t)i);
  uintx2 o;
  o.x = (unsigned)f32_to_bf16(f.x) | ((unsigned)f32_to_bf16(f.y) << 16);
  o.y = (unsigned)f32_to_bf16(f.z) | ((unsigned)f32_to_bf16(f.w) << 16);
  *(uintx2*)(out + 4 * (size_t)i) = o;
}

// ---------------------------------------------------------------- GEMM: C = (A @ W^T + bias) * scale
// A: [M,1024] bf16 row-major. W: [1024,1024] bf16 row-major (acts as B^T).
// MODE 0: bf16 out, [B,H,S,Dh].  MODE 1: fp32 out, row-major.  MODE 2: bf16 out, [B,H,Dh,S].
template <int MODE>
__global__ __launch_bounds__(256)
void gemm_bf16_kernel(const u16* __restrict__ A, const u16* __restrict__ W,
                      const float* __restrict__ bias, float scale,
                      u16* __restrict__ outb, float* __restrict__ outf) {
  __shared__ __attribute__((aligned(16))) u16 As[2][128 * 40];  // 128x32, stride 40 (80B)
  __shared__ __attribute__((aligned(16))) u16 Bs[2][128 * 40];

  const int tid  = threadIdx.x;
  const int lane = tid & 31, wave = tid >> 5;
  const int l16 = lane & 15, lh = lane >> 4;
  const int wm = wave & 1, wn = wave >> 1;          // 2x4 wave grid
  const int mBase = blockIdx.y * 128;
  const int nBase = blockIdx.x * 128;

  floatx8 acc[4][2] = {};

  const int ldr = tid >> 1;          // 0..127
  const int ldc = (tid & 1) * 16;    // 0 or 16 elements (32B half)
  const u16* gA = A + (size_t)(mBase + ldr) * 1024 + ldc;
  const u16* gW = W + (size_t)(nBase + ldr) * 1024 + ldc;
  const int sOff = ldr * 40 + ldc;

  // issue async copy of k-tile 0 into buffer 0
  async_copy_b128(&As[0][sOff],     gA);
  async_copy_b128(&As[0][sOff + 8], gA + 8);
  async_copy_b128(&Bs[0][sOff],     gW);
  async_copy_b128(&Bs[0][sOff + 8], gW + 8);

#pragma unroll
  for (int k0 = 0; k0 < 1024; k0 += 32) {
    const int cur = (k0 >> 5) & 1, nxt = cur ^ 1;   // compile-time after unroll
    wait_async0();                         // my async copies into buf[cur] done
    __syncthreads();                       // everyone's copies visible; buf[nxt] free
    if (k0 + 32 < 1024) {                  // overlap next tile copy with compute
      async_copy_b128(&As[nxt][sOff],     gA + k0 + 32);
      async_copy_b128(&As[nxt][sOff + 8], gA + k0 + 40);
      async_copy_b128(&Bs[nxt][sOff],     gW + k0 + 32);
      async_copy_b128(&Bs[nxt][sOff + 8], gW + k0 + 40);
    }
    if (k0 + 64 < 1024) {                  // global_prefetch_b8 two tiles ahead
      __builtin_prefetch(gA + k0 + 64, 0, 1);
      __builtin_prefetch(gW + k0 + 64, 0, 1);
    }

    Frag32B af[4], bf_[2];
#pragma unroll
    for (int i = 0; i < 4; i++) {          // A-frag: K = 8*lh..+7 and +16..+23
      const u16* p = &As[cur][(wm * 64 + i * 16 + l16) * 40 + lh * 8];
      af[i].q[0] = *(const uintx4*)p;
      af[i].q[1] = *(const uintx4*)(p + 16);
    }
#pragma unroll
    for (int j = 0; j < 2; j++) {          // B-frag: col n, K = 8*lh..+7 and +16..+23
      const u16* p = &Bs[cur][(wn * 32 + j * 16 + l16) * 40 + lh * 8];
      bf_[j].q[0] = *(const uintx4*)p;
      bf_[j].q[1] = *(const uintx4*)(p + 16);
    }
#pragma unroll
    for (int i = 0; i < 4; i++)
#pragma unroll
      for (int j = 0; j < 2; j++)
        acc[i][j] = __builtin_amdgcn_wmma_f32_16x16x32_bf16(
            false, af[i].v, false, bf_[j].v, (short)0, acc[i][j], false, false);
  }

#pragma unroll
  for (int j = 0; j < 2; j++) {
    const int n = nBase + wn * 32 + j * 16 + l16;
    const float bv = bias[n];
#pragma unroll
    for (int i = 0; i < 4; i++) {
#pragma unroll
      for (int r = 0; r < 8; r++) {
        const int m = mBase + wm * 64 + i * 16 + r + 8 * lh;
        const float v = (acc[i][j][r] + bv) * scale;
        if (MODE == 0) {
          const int b = m >> 11, s = m & 2047, h = n >> 6, dh = n & 63;
          outb[(((size_t)(b * HEADS + h) * SEQ + s) << 6) + dh] = f32_to_bf16(v);
        } else if (MODE == 2) {
          const int b = m >> 11, s = m & 2047, h = n >> 6, dh = n & 63;
          outb[((size_t)((b * HEADS + h) * 64 + dh) << 11) + s] = f32_to_bf16(v);
        } else {
          outf[(size_t)m * 1024 + n] = v;
        }
      }
    }
  }
}

// ---------------------------------------------------------------- flash attention
// Q (pre-scaled by 1/8), K bf16 [B*H, S, 64]; Vt bf16 [B*H, 64, S]; O bf16 [B, S, 1024].
// Double-buffered K/V staging via async-to-LDS; P staging overlays the Q region.
__global__ __launch_bounds__(256)
void flash_attn_kernel(const u16* __restrict__ Qp, const u16* __restrict__ Kp,
                       const u16* __restrict__ Vtg, u16* __restrict__ O) {
  __shared__ __attribute__((aligned(16))) u16 Qs[128 * 72];     // reused as P staging
  __shared__ __attribute__((aligned(16))) u16 Ks[2][64 * 72];
  __shared__ __attribute__((aligned(16))) u16 Vt[2][64 * 72];   // [dh][key]

  const int tid  = threadIdx.x;
  const int lane = tid & 31, wave = tid >> 5;
  const int l16 = lane & 15, lh = lane >> 4;
  const int qbase = blockIdx.x * 128;
  const int bh = blockIdx.y;
  const int b = bh >> 4, h = bh & 15;
  const size_t headOff = (size_t)bh * SEQ * 64;

  const int kr = tid >> 2, kpart = (tid & 3) * 16;   // K/V staging mapping
  const u16* gK0 = Kp + headOff + (size_t)kr * 64 + kpart;               // + key*64
  const u16* gV0 = Vtg + ((size_t)(bh * 64 + kr) << 11) + kpart;         // + key

  // stage Q block (128 x 64), build A-fragments, then the region becomes P staging
  {
    const int r = tid >> 1, half = (tid & 1) * 32;
    const u16* gq = Qp + headOff + (size_t)(qbase + r) * 64 + half;
#pragma unroll
    for (int q = 0; q < 4; q++)
      *(uintx4*)&Qs[r * 72 + half + 8 * q] = *(const uintx4*)(gq + 8 * q);
  }
  // issue async K/V copies for key block 0 into buffer 0
  async_copy_b128(&Ks[0][kr * 72 + kpart],     gK0);
  async_copy_b128(&Ks[0][kr * 72 + kpart + 8], gK0 + 8);
  async_copy_b128(&Vt[0][kr * 72 + kpart],     gV0);
  async_copy_b128(&Vt[0][kr * 72 + kpart + 8], gV0 + 8);
  __syncthreads();

  Frag32B qf[2];
#pragma unroll
  for (int c = 0; c < 2; c++) {
    const u16* p = &Qs[(wave * 16 + l16) * 72 + c * 32 + lh * 8];
    qf[c].q[0] = *(const uintx4*)p;
    qf[c].q[1] = *(const uintx4*)(p + 16);
  }

  floatx8 oacc[4] = {};
  float runmax[8], runsum[8];
#pragma unroll
  for (int r = 0; r < 8; r++) { runmax[r] = -1e30f; runsum[r] = 0.0f; }

  const int nkb = (qbase >> 6) + 2;   // causal key blocks; always even
  for (int kbase = 0; kbase < nkb; kbase += 2) {
#pragma unroll
    for (int half = 0; half < 2; half++) {          // compile-time buffer index
      const int kb = kbase + half;
      const int keyb = kb * 64;
      const int cur = half, nxt = half ^ 1;
      wait_async0();                       // my copies into buf[cur] complete
      __syncthreads();                     // all waves' copies visible; buf[nxt] free
      if (kb + 1 < nkb) {                  // issue next key block while computing
        const u16* gk = gK0 + (size_t)(keyb + 64) * 64;
        const u16* gv = gV0 + keyb + 64;
        async_copy_b128(&Ks[nxt][kr * 72 + kpart],     gk);
        async_copy_b128(&Ks[nxt][kr * 72 + kpart + 8], gk + 8);
        async_copy_b128(&Vt[nxt][kr * 72 + kpart],     gv);
        async_copy_b128(&Vt[nxt][kr * 72 + kpart + 8], gv + 8);
        if (kb + 2 < nkb) {
          __builtin_prefetch(gK0 + (size_t)(keyb + 128) * 64, 0, 1);
          __builtin_prefetch(gV0 + keyb + 128, 0, 1);
        }
      }

      // scores = Q @ K^T  (M=16 q rows, N=64 keys, K=64 dh); Q pre-scaled by 1/8
      floatx8 sc[4];
#pragma unroll
      for (int jn = 0; jn < 4; jn++)
#pragma unroll
        for (int r = 0; r < 8; r++) sc[jn][r] = 0.0f;
#pragma unroll
      for (int jn = 0; jn < 4; jn++) {
#pragma unroll
        for (int c = 0; c < 2; c++) {
          Frag32B kf;
          const u16* p = &Ks[cur][(jn * 16 + l16) * 72 + c * 32 + lh * 8];
          kf.q[0] = *(const uintx4*)p;
          kf.q[1] = *(const uintx4*)(p + 16);
          sc[jn] = __builtin_amdgcn_wmma_f32_16x16x32_bf16(
              false, qf[c].v, false, kf.v, (short)0, sc[jn], false, false);
        }
      }

      // causal mask + online softmax (row stats via half-wave butterflies)
      u16* pw = &Qs[wave * 16 * 72];      // per-wave P staging (Q region reused)
#pragma unroll
      for (int r = 0; r < 8; r++) {
        const int qg = qbase + wave * 16 + r + 8 * lh;
        float mx = -1e30f;
#pragma unroll
        for (int jn = 0; jn < 4; jn++) {
          const int kg = keyb + jn * 16 + l16;
          float v = sc[jn][r];
          v = (kg > qg) ? -1e30f : v;
          sc[jn][r] = v;
          mx = fmaxf(mx, v);
        }
        for (int msk = 1; msk < 16; msk <<= 1) mx = fmaxf(mx, __shfl_xor(mx, msk, 32));
        const float nm = fmaxf(runmax[r], mx);
        const float so = __expf(runmax[r] - nm);
        runmax[r] = nm;
        float rs = 0.0f;
#pragma unroll
        for (int jn = 0; jn < 4; jn++) {
          const float pv = __expf(sc[jn][r] - nm);
          sc[jn][r] = pv;
          rs += pv;
        }
        for (int msk = 1; msk < 16; msk <<= 1) rs += __shfl_xor(rs, msk, 32);
        runsum[r] = runsum[r] * so + rs;
#pragma unroll
        for (int jd = 0; jd < 4; jd++) oacc[jd][r] *= so;
      }

      // P (C-layout f32) -> LDS bf16 -> A-fragments
#pragma unroll
      for (int jn = 0; jn < 4; jn++)
#pragma unroll
        for (int r = 0; r < 8; r++)
          pw[(r + 8 * lh) * 72 + jn * 16 + l16] = f32_to_bf16(sc[jn][r]);

      Frag32B pf[2];
#pragma unroll
      for (int c = 0; c < 2; c++) {
        const u16* p = &pw[l16 * 72 + c * 32 + lh * 8];
        pf[c].q[0] = *(const uintx4*)p;
        pf[c].q[1] = *(const uintx4*)(p + 16);
      }

      // O += P @ V  (K=64 keys via two 32-chunks; B-frag from transposed V)
#pragma unroll
      for (int jd = 0; jd < 4; jd++) {
#pragma unroll
        for (int c = 0; c < 2; c++) {
          Frag32B vf;
          const u16* p = &Vt[cur][(jd * 16 + l16) * 72 + c * 32 + lh * 8];
          vf.q[0] = *(const uintx4*)p;
          vf.q[1] = *(const uintx4*)(p + 16);
          oacc[jd] = __builtin_amdgcn_wmma_f32_16x16x32_bf16(
              false, pf[c].v, false, vf.v, (short)0, oacc[jd], false, false);
        }
      }
    }
  }

  // normalize + write attention output as bf16 [B,S,1024]
#pragma unroll
  for (int r = 0; r < 8; r++) {
    const float inv = 1.0f / runsum[r];
    const int sq = qbase + wave * 16 + r + 8 * lh;
#pragma unroll
    for (int jd = 0; jd < 4; jd++) {
      O[((size_t)(b * SEQ + sq) << 10) + h * 64 + jd * 16 + l16] =
          f32_to_bf16(oacc[jd][r] * inv);
    }
  }
}

// ---------------------------------------------------------------- launcher
extern "C" void kernel_launch(void* const* d_in, const int* in_sizes, int n_in,
                              void* d_out, int out_size, void* d_ws, size_t ws_size,
                              hipStream_t stream) {
  const float* x  = (const float*)d_in[0];
  const float* Wq = (const float*)d_in[1];
  const float* bq = (const float*)d_in[2];
  const float* Wk = (const float*)d_in[3];
  const float* bk = (const float*)d_in[4];
  const float* Wv = (const float*)d_in[5];
  const float* bv = (const float*)d_in[6];
  const float* Wo = (const float*)d_in[7];
  const float* bo = (const float*)d_in[8];

  char* ws = (char*)d_ws;
  u16* xb  = (u16*)(ws);                         // 8 MB  x bf16
  u16* wqb = (u16*)(ws + ((size_t)8  << 20));    // 2 MB each
  u16* wkb = (u16*)(ws + ((size_t)10 << 20));
  u16* wvb = (u16*)(ws + ((size_t)12 << 20));
  u16* wob = (u16*)(ws + ((size_t)14 << 20));
  u16* Qb  = (u16*)(ws + ((size_t)16 << 20));    // 8 MB, [B,H,S,Dh], pre-scaled 1/8
  u16* Kb  = (u16*)(ws + ((size_t)24 << 20));    // 8 MB, [B,H,S,Dh]
  u16* Vtb = (u16*)(ws + ((size_t)32 << 20));    // 8 MB, [B,H,Dh,S] (transposed)
  u16* Ob  = (u16*)(ws + ((size_t)40 << 20));    // 8 MB attn out [B,S,D]

  // pack to bf16
  cvt_f32_bf16_kernel<<<(M_TOT * D_MODEL / 4) / 256, 256, 0, stream>>>(x, xb, M_TOT * D_MODEL / 4);
  cvt_f32_bf16_kernel<<<(D_MODEL * D_MODEL / 4) / 256, 256, 0, stream>>>(Wq, wqb, D_MODEL * D_MODEL / 4);
  cvt_f32_bf16_kernel<<<(D_MODEL * D_MODEL / 4) / 256, 256, 0, stream>>>(Wk, wkb, D_MODEL * D_MODEL / 4);
  cvt_f32_bf16_kernel<<<(D_MODEL * D_MODEL / 4) / 256, 256, 0, stream>>>(Wv, wvb, D_MODEL * D_MODEL / 4);
  cvt_f32_bf16_kernel<<<(D_MODEL * D_MODEL / 4) / 256, 256, 0, stream>>>(Wo, wob, D_MODEL * D_MODEL / 4);

  // QKV projections (WMMA GEMM); Q pre-scaled by 1/sqrt(Dh); V written pre-transposed
  dim3 gg(D_MODEL / 128, M_TOT / 128);
  gemm_bf16_kernel<0><<<gg, 256, 0, stream>>>(xb, wqb, bq, 0.125f, Qb, nullptr);
  gemm_bf16_kernel<0><<<gg, 256, 0, stream>>>(xb, wkb, bk, 1.0f,   Kb, nullptr);
  gemm_bf16_kernel<2><<<gg, 256, 0, stream>>>(xb, wvb, bv, 1.0f,   Vtb, nullptr);

  // causal flash attention (WMMA for QK^T and P·V)
  flash_attn_kernel<<<dim3(SEQ / 128, BATCH * HEADS), 256, 0, stream>>>(Qb, Kb, Vtb, Ob);

  // output projection -> fp32 d_out
  gemm_bf16_kernel<1><<<gg, 256, 0, stream>>>(Ob, wob, bo, 1.0f, nullptr, (float*)d_out);
}
